// DeepBiLSTMGNN_36876589204179
// MI455X (gfx1250) — matrix operand order repaired and checked by reading, not compiled, not run on hardware
//
#include <hip/hip_runtime.h>
#include <hip/hip_bf16.h>
#include <cstdint>

// MI455X / gfx1250, wave32.
// Matmuls: v_wmma_f32_16x16x32_bf16 (f32 accumulate).
// Data movement: global_load_async_to_lds_b128 (ASYNCcnt) for x/adj staging,
// tensor_load_to_lds (TDM, TENSORcnt) for LSTM weight staging.
// Shapes fixed per reference: B=4, S=64, N=256, F=32, H=64.

typedef __attribute__((ext_vector_type(16))) __bf16 v16bf;
typedef __attribute__((ext_vector_type(8)))  __bf16 v8bf;
typedef __attribute__((ext_vector_type(8)))  float  v8f;
typedef __attribute__((ext_vector_type(4)))  unsigned u32x4;
typedef __attribute__((ext_vector_type(8)))  unsigned u32x8;

static __device__ __forceinline__ v16bf ld16(const __bf16* p0, const __bf16* p1) {
    v8bf a = *(const v8bf*)p0;
    v8bf b = *(const v8bf*)p1;
    return __builtin_shufflevector(a, b, 0,1,2,3,4,5,6,7,8,9,10,11,12,13,14,15);
}

static __device__ __forceinline__ v8f wmma_bf16(v16bf a, v16bf b, v8f c) {
    return __builtin_amdgcn_wmma_f32_16x16x32_bf16(false, a, false, b, (short)0, c, false, false);
}

// 16-byte global -> LDS async copy (per-lane), tracked by ASYNCcnt.
static __device__ __forceinline__ void async_ld16(unsigned lds_off, const void* gptr) {
    asm volatile("global_load_async_to_lds_b128 %0, %1, off"
                 :: "v"(lds_off), "v"((unsigned long long)(uintptr_t)gptr)
                 : "memory");
}
static __device__ __forceinline__ void wait_async_le8() {
    asm volatile("s_wait_asynccnt 0x8" ::: "memory");
}
static __device__ __forceinline__ void wait_async_0() {
    asm volatile("s_wait_asynccnt 0x0" ::: "memory");
}

static __device__ __forceinline__ float wred_max(float v) {
    #pragma unroll
    for (int m = 16; m >= 1; m >>= 1) v = fmaxf(v, __shfl_xor(v, m, 32));
    return v;
}
static __device__ __forceinline__ float wred_sum(float v) {
    #pragma unroll
    for (int m = 16; m >= 1; m >>= 1) v += __shfl_xor(v, m, 32);
    return v;
}
static __device__ __forceinline__ float sigf(float x) { return 1.f / (1.f + __expf(-x)); }

// ---------------------------------------------------------------------------
// GAT layer: one block per graph (M = B*S = 256 graphs, N = 256 nodes, H = 64)
// ---------------------------------------------------------------------------
template <int FIN>
__global__ void __launch_bounds__(256)
gat_kernel(const float* __restrict__ X,     // [256][256][FIN]
           const float* __restrict__ ADJ,   // [256][256][256]
           const float* __restrict__ W,     // [FIN][64]
           const float* __restrict__ avec,  // [128]
           const float* __restrict__ gamma, // [64]
           const float* __restrict__ beta,  // [64]
           float* __restrict__ OUT)         // [256][256][64]
{
    extern __shared__ char smem[];
    constexpr int XF_OFF  = 0;                         // f32 x staging (async dst)
    constexpr int XB_OFF  = XF_OFF  + 256 * FIN * 4;   // bf16 x (A operand)
    constexpr int WT_OFF  = XB_OFF  + 256 * FIN * 2;   // bf16 W^T (B operand)
    constexpr int HT_OFF  = WT_OFF  + 64 * FIN * 2;    // bf16 h^T [64][256]
    constexpr int ATT_OFF = HT_OFF  + 32768;           // bf16 att strip [32][256]
    constexpr int SI_OFF  = ATT_OFF + 16384;           // f32 s_i [256]
    constexpr int SJ_OFF  = SI_OFF  + 1024;            // f32 s_j [256]
    constexpr int HP_OFF  = SJ_OFF  + 1024;            // f32 hp strip [32][64]
    constexpr int ADJ_OFF = HP_OFF  + 8192;            // f32 adj strips [2][32][256]

    float*  xf  = (float*) (smem + XF_OFF);
    __bf16* xb  = (__bf16*)(smem + XB_OFF);
    __bf16* wT  = (__bf16*)(smem + WT_OFF);
    __bf16* hT  = (__bf16*)(smem + HT_OFF);
    __bf16* att = (__bf16*)(smem + ATT_OFF);
    float*  sI  = (float*) (smem + SI_OFF);
    float*  sJ  = (float*) (smem + SJ_OFF);
    float*  hp  = (float*) (smem + HP_OFF);
    float*  adjb= (float*) (smem + ADJ_OFF);

    const unsigned dynbase = __builtin_amdgcn_groupstaticsize();
    const int m    = blockIdx.x;
    const int tid  = threadIdx.x;
    const int wave = tid >> 5;
    const int lane = tid & 31;
    const int half = lane >> 4;
    const int l15  = lane & 15;

    // ---- async-stage x tile (f32) and adj strip 0 ----
    const float* Xm = X + (size_t)m * 256 * FIN;
    {
        constexpr int XCHUNKS = 256 * FIN * 4 / 16;     // 16B chunks
        #pragma unroll
        for (int c = tid; c < XCHUNKS; c += 256)
            async_ld16(dynbase + XF_OFF + c * 16, (const char*)Xm + c * 16);
        const float* arow = ADJ + (size_t)m * 256 * 256;
        for (int c = tid; c < 2048; c += 256)           // 32 KB strip 0
            async_ld16(dynbase + ADJ_OFF + c * 16, (const char*)arow + c * 16);
    }

    // ---- convert W^T while async copies fly ----
    for (int i = tid; i < 64 * FIN; i += 256) {
        int col = i / FIN, f = i - col * FIN;
        wT[i] = (__bf16)W[f * 64 + col];
    }
    wait_async_le8();          // retire the x-tile chunks (adj strip 0 may still fly)
    __syncthreads();
    for (int i = tid; i < 256 * FIN; i += 256) xb[i] = (__bf16)xf[i];
    __syncthreads();

    // ---- h = x @ W  (WMMA, write h^T bf16) ----
    for (int tile = wave; tile < 64; tile += 8) {
        int mt = tile >> 2, nt = tile & 3;
        v8f acc = {};
        #pragma unroll
        for (int kk = 0; kk < FIN; kk += 32) {
            int row = mt * 16 + l15;
            int ka  = kk + half * 8;
            v16bf A = ld16(&xb[row * FIN + ka], &xb[row * FIN + ka + 16]);
            int col = nt * 16 + l15;
            int kb  = kk + half * 8;
            v16bf Bv = ld16(&wT[col * FIN + kb], &wT[col * FIN + kb + 16]);
            acc = wmma_bf16(A, Bv, acc);
        }
        int col = nt * 16 + l15;
        #pragma unroll
        for (int r = 0; r < 8; ++r) {
            int row = mt * 16 + r + 8 * half;
            hT[col * 256 + row] = (__bf16)acc[r];
        }
    }
    __syncthreads();

    // ---- attention logits: s_i = h.a1, s_j = h.a2 ----
    for (int i = tid; i < 256; i += 256) {
        float si = 0.f, sj = 0.f;
        #pragma unroll 4
        for (int n = 0; n < 64; ++n) {
            float hv = (float)hT[n * 256 + i];
            si += hv * avec[n];
            sj += hv * avec[64 + n];
        }
        sI[i] = si; sJ[i] = sj;
    }
    __syncthreads();

    // ---- per 32-row strip: masked softmax -> WMMA aggregation -> LN+ELU ----
    for (int rb = 0; rb < 8; ++rb) {
        // prefetch next adj strip into the other LDS buffer (overlaps compute)
        if (rb < 7) {
            const float* arow = ADJ + ((size_t)m * 256 + (rb + 1) * 32) * 256;
            unsigned dst = dynbase + ADJ_OFF + ((rb + 1) & 1) * 32768;
            for (int c = tid; c < 2048; c += 256)
                async_ld16(dst + c * 16, (const char*)arow + c * 16);
            wait_async_le8();      // strip rb resident; strip rb+1 in flight
        } else {
            wait_async_0();
        }
        __syncthreads();
        const float* adjf = adjb + (rb & 1) * 8192;

        // softmax: each wave owns 4 rows
        for (int q = 0; q < 4; ++q) {
            int rl = wave * 4 + q;
            int i  = rb * 32 + rl;
            float ev[8];
            float mx = -3.0e38f;
            float si = sI[i];
            #pragma unroll
            for (int k = 0; k < 8; ++k) {
                int j = lane + k * 32;
                float s = si + sJ[j];
                float e = s > 0.f ? s : 0.2f * s;
                e = adjf[rl * 256 + j] > 0.f ? e : -9.0e15f;
                ev[k] = e;
                mx = fmaxf(mx, e);
            }
            mx = wred_max(mx);
            float sum = 0.f;
            #pragma unroll
            for (int k = 0; k < 8; ++k) {
                float p = __expf(ev[k] - mx);
                ev[k] = p;
                sum += p;
            }
            sum = wred_sum(sum);
            float inv = 1.f / sum;
            #pragma unroll
            for (int k = 0; k < 8; ++k) {
                int j = lane + k * 32;
                att[rl * 256 + j] = (__bf16)(ev[k] * inv);
            }
        }
        __syncthreads();

        // hp[32][64] = att[32][256] @ h[256][64]; 8 tiles, one per wave
        {
            int mt = wave >> 2, nt = wave & 3;
            v8f acc = {};
            #pragma unroll
            for (int kk = 0; kk < 256; kk += 32) {
                int row = mt * 16 + l15;
                int ka  = kk + half * 8;
                v16bf A = ld16(&att[row * 256 + ka], &att[row * 256 + ka + 16]);
                int col = nt * 16 + l15;
                int kb  = kk + half * 8;
                v16bf Bv = ld16(&hT[col * 256 + kb], &hT[col * 256 + kb + 16]);
                acc = wmma_bf16(A, Bv, acc);
            }
            int col = nt * 16 + l15;
            #pragma unroll
            for (int r = 0; r < 8; ++r) {
                int row = mt * 16 + r + 8 * half;
                hp[row * 64 + col] = acc[r];
            }
        }
        __syncthreads();

        // LayerNorm over H=64 + ELU, store f32
        if (tid < 32) {
            int grow = rb * 32 + tid;
            float s = 0.f, s2 = 0.f;
            #pragma unroll 4
            for (int h = 0; h < 64; ++h) {
                float v = hp[tid * 64 + h];
                s += v; s2 += v * v;
            }
            float mu  = s * (1.f / 64.f);
            float var = s2 * (1.f / 64.f) - mu * mu;
            float rin = rsqrtf(var + 1e-5f);
            float* outp = OUT + ((size_t)m * 256 + grow) * 64;
            #pragma unroll 4
            for (int h = 0; h < 64; ++h) {
                float v = (hp[tid * 64 + h] - mu) * rin * gamma[h] + beta[h];
                outp[h] = v > 0.f ? v : (__expf(v) - 1.f);
            }
        }
        __syncthreads();
    }
}

// ---------------------------------------------------------------------------
// Residual + gate combine, transpose (B,S,N,H)->(B,N,S,H), emit bf16 sequence
// ---------------------------------------------------------------------------
__global__ void __launch_bounds__(256)
comb_kernel(const float* __restrict__ X,     // [4][64][256][32]
            const float* __restrict__ G,     // [256][256][64]
            const float* __restrict__ resW,  // [32][64]
            const float* __restrict__ resb,  // [64]
            const float* __restrict__ alpha, // [1]
            __bf16* __restrict__ seq)        // [1024][64][64] bf16
{
    int idx = blockIdx.x * 256 + threadIdx.x;       // < 2^22
    int h = idx & 63;
    int n = (idx >> 6) & 255;
    int s = (idx >> 14) & 63;
    int b = idx >> 20;
    const float* xr = X + (((size_t)(b * 64 + s) * 256) + n) * 32;
    float acc = resb[h];
    #pragma unroll 8
    for (int f = 0; f < 32; ++f) acc += xr[f] * resW[f * 64 + h];
    float gate = fminf(fmaxf(alpha[0], 0.f), 1.f);
    acc += gate * G[(((size_t)(b * 64 + s) * 256) + n) * 64 + h];
    seq[(((size_t)(b * 256 + n) * 64) + s) * 64 + h] = (__bf16)acc;
}

// ---------------------------------------------------------------------------
// One BiLSTM layer: grid.x = R/16 row tiles, grid.y = direction.
// Wih staged via TDM (tensor_load_to_lds), gate GEMM via WMMA.
// ---------------------------------------------------------------------------
template <int DIN>
__global__ void __launch_bounds__(256)
lstm_kernel(const __bf16* __restrict__ Xin,   // [1024][64][DIN] bf16
            const float* __restrict__ Wih,    // [2][DIN][256]
            const float* __restrict__ Whh,    // [2][64][256]
            const float* __restrict__ bih,    // [2][256]
            const float* __restrict__ bhh,    // [2][256]
            __bf16* __restrict__ obuf,        // [1024][64][128] bf16 or null
            float* __restrict__ hlast)        // [2][1024][64] or null
{
    extern __shared__ char smem[];
    constexpr int WF_OFF   = 0;                        // f32 Wih staging (TDM dst)
    constexpr int WI_OFF   = WF_OFF  + 256 * DIN * 4;  // bf16 Wih^T [256][DIN]
    constexpr int WH_OFF   = WI_OFF  + 256 * DIN * 2;  // bf16 Whh^T [256][64]
    constexpr int HB_OFF   = WH_OFF  + 32768;          // bf16 h state [16][64]
    constexpr int BIAS_OFF = HB_OFF  + 2048;           // f32 bias [256]
    constexpr int CST_OFF  = BIAS_OFF+ 1024;           // f32 cell [16][64]
    constexpr int GST_OFF  = CST_OFF + 4096;           // f32 gates [16][256]

    float*  wf   = (float*) (smem + WF_OFF);
    __bf16* wiT  = (__bf16*)(smem + WI_OFF);
    __bf16* whT  = (__bf16*)(smem + WH_OFF);
    __bf16* hb   = (__bf16*)(smem + HB_OFF);
    float*  bias = (float*) (smem + BIAS_OFF);
    float*  cst  = (float*) (smem + CST_OFF);
    float*  gst  = (float*) (smem + GST_OFF);

    const unsigned dynbase = __builtin_amdgcn_groupstaticsize();
    const int tid  = threadIdx.x;
    const int wave = tid >> 5;
    const int lane = tid & 31;
    const int half = lane >> 4;
    const int l15  = lane & 15;
    const int dir  = blockIdx.y;
    const int r0   = blockIdx.x * 16;

    const float* Wd = Wih + (size_t)dir * DIN * 256;
    const float* Ud = Whh + (size_t)dir * 64 * 256;

    // ---- TDM: DMA Wih (f32, DIN*256 elements) into LDS staging; wave 0 only ----
    if (tid < 32) {
        unsigned long long ga = (unsigned long long)(uintptr_t)Wd;
        unsigned ga_lo = (unsigned)__builtin_amdgcn_readfirstlane((int)(unsigned)ga);
        unsigned ga_hi = (unsigned)__builtin_amdgcn_readfirstlane((int)(unsigned)(ga >> 32));
        unsigned lds_a = (unsigned)__builtin_amdgcn_readfirstlane((int)(dynbase + WF_OFF));
        unsigned dim0  = (unsigned)(DIN * 256);          // 1-D tile, data_size=4B
        u32x4 g0 = { 1u,                                 // count=1, is_restore=0
                     lds_a,                              // lds_addr
                     ga_lo,                              // global_addr[31:0]
                     (ga_hi & 0x01ffffffu) | 0x80000000u };  // addr[56:32] | type=2
        u32x8 g1 = { 0x00020000u,                        // wg_mask=0, data_size=2(4B)
                     (dim0 & 0xffffu) << 16,             // tensor_dim0[15:0]
                     (dim0 >> 16) | (1u << 16),          // tensor_dim0[31:16], dim1=1
                     (dim0 & 0xffffu) << 16,             // tile_dim0
                     0u,                                 // tile_dim1/2 = 0
                     dim0,                               // tensor_dim0_stride[31:0]
                     0u, 0u };
        asm volatile("tensor_load_to_lds %0, %1" :: "s"(g0), "s"(g1) : "memory");
        __builtin_amdgcn_s_wait_tensorcnt(0);
    }

    // ---- Whh^T / bias conversion overlaps the TDM ----
    for (int i = tid; i < 256 * 64; i += 256) {
        int u = i >> 6, d = i & 63;
        whT[i] = (__bf16)Ud[d * 256 + u];
    }
    for (int i = tid; i < 256; i += 256) bias[i] = bih[dir * 256 + i] + bhh[dir * 256 + i];
    for (int i = tid; i < 1024; i += 256) { hb[i] = (__bf16)0.f; cst[i] = 0.f; }
    __syncthreads();                       // wave0 passed s_wait_tensorcnt -> wf valid

    for (int i = tid; i < 256 * DIN; i += 256) {
        int u = i / DIN, d = i - u * DIN;
        wiT[i] = (__bf16)wf[d * 256 + u];
    }
    __syncthreads();

    for (int s = 0; s < 64; ++s) {
        const int t = (dir != 0) ? (63 - s) : s;

        // gates[16][256] = x_t @ Wih + h @ Whh + bias   (WMMA)
        #pragma unroll
        for (int it = 0; it < 2; ++it) {
            const int nt = wave + it * 8;
            v8f acc = {};
            const __bf16* xrow = Xin + ((size_t)(r0 + l15) * 64 + t) * DIN;
            const int u = nt * 16 + l15;
            #pragma unroll
            for (int kk = 0; kk < DIN; kk += 32) {
                int ka = kk + half * 8;
                v16bf A = ld16(xrow + ka, xrow + ka + 16);
                int kb = kk + half * 8;
                v16bf Bv = ld16(&wiT[u * DIN + kb], &wiT[u * DIN + kb + 16]);
                acc = wmma_bf16(A, Bv, acc);
            }
            #pragma unroll
            for (int kk = 0; kk < 64; kk += 32) {
                int ka = kk + half * 8;
                v16bf A = ld16(&hb[l15 * 64 + ka], &hb[l15 * 64 + ka + 16]);
                int kb = kk + half * 8;
                v16bf Bv = ld16(&whT[u * 64 + kb], &whT[u * 64 + kb + 16]);
                acc = wmma_bf16(A, Bv, acc);
            }
            #pragma unroll
            for (int r = 0; r < 8; ++r) {
                int row = r + 8 * half;
                gst[row * 256 + u] = acc[r] + bias[u];
            }
        }
        __syncthreads();

        // elementwise c/h update (torch gate order i,f,g,o)
        for (int idx = tid; idx < 1024; idx += 256) {
            int row = idx >> 6, u = idx & 63;
            float iv = gst[row * 256 + u];
            float fv = gst[row * 256 + 64 + u];
            float gv = gst[row * 256 + 128 + u];
            float ov = gst[row * 256 + 192 + u];
            float c  = sigf(fv) * cst[idx] + sigf(iv) * tanhf(gv);
            float hv = sigf(ov) * tanhf(c);
            cst[idx] = c;
            hb[idx]  = (__bf16)hv;
            if (obuf)
                obuf[((size_t)(r0 + row) * 64 + t) * 128 + dir * 64 + u] = (__bf16)hv;
        }
        __syncthreads();
    }

    if (hlast) {
        for (int idx = tid; idx < 1024; idx += 256) {
            int row = idx >> 6, u = idx & 63;
            hlast[((size_t)dir * 1024 + (r0 + row)) * 64 + u] = (float)hb[idx];
        }
    }
}

// ---------------------------------------------------------------------------
// Final FC: pred[r] = [hf, hb] @ fc_W + fc_b
// ---------------------------------------------------------------------------
__global__ void __launch_bounds__(256)
fc_kernel(const float* __restrict__ hlast,  // [2][1024][64]
          const float* __restrict__ fcW,    // [128]
          const float* __restrict__ fcb,    // [1]
          float* __restrict__ out)          // [1024]
{
    int r = blockIdx.x * 256 + threadIdx.x;
    if (r >= 1024) return;
    float acc = fcb[0];
    const float* hf  = hlast + (size_t)r * 64;
    const float* hbp = hlast + (size_t)1024 * 64 + (size_t)r * 64;
    #pragma unroll 8
    for (int k = 0; k < 64; ++k) acc += hf[k] * fcW[k] + hbp[k] * fcW[64 + k];
    out[r] = acc;
}

// ---------------------------------------------------------------------------
extern "C" void kernel_launch(void* const* d_in, const int* in_sizes, int n_in,
                              void* d_out, int out_size, void* d_ws, size_t ws_size,
                              hipStream_t stream) {
    const float* x       = (const float*)d_in[0];
    const float* adj     = (const float*)d_in[1];
    const float* gat0_W  = (const float*)d_in[2];
    const float* gat0_a  = (const float*)d_in[3];
    const float* gat0_g  = (const float*)d_in[4];
    const float* gat0_b  = (const float*)d_in[5];
    const float* gat1_W  = (const float*)d_in[6];
    const float* gat1_a  = (const float*)d_in[7];
    const float* gat1_g  = (const float*)d_in[8];
    const float* gat1_b  = (const float*)d_in[9];
    const float* res_W   = (const float*)d_in[10];
    const float* res_b   = (const float*)d_in[11];
    const float* alpha   = (const float*)d_in[12];
    const float* l0_Wih  = (const float*)d_in[13];
    const float* l0_Whh  = (const float*)d_in[14];
    const float* l0_bih  = (const float*)d_in[15];
    const float* l0_bhh  = (const float*)d_in[16];
    const float* l1_Wih  = (const float*)d_in[17];
    const float* l1_Whh  = (const float*)d_in[18];
    const float* l1_bih  = (const float*)d_in[19];
    const float* l1_bhh  = (const float*)d_in[20];
    const float* fc_W    = (const float*)d_in[21];
    const float* fc_b    = (const float*)d_in[22];

    char* ws = (char*)d_ws;
    size_t off = 0;
    float*  g1    = (float*)(ws + off);  off += (size_t)256 * 256 * 64 * 4;  // 16 MB
    float*  g2    = (float*)(ws + off);  off += (size_t)256 * 256 * 64 * 4;  // 16 MB
    __bf16* seqbf = (__bf16*)(ws + off); off += (size_t)1024 * 64 * 64 * 2;  // 8 MB
    __bf16* o1bf  = (__bf16*)(ws + off); off += (size_t)1024 * 64 * 128 * 2; // 16 MB
    float*  hlast = (float*)(ws + off);  off += (size_t)2 * 1024 * 64 * 4;   // 0.5 MB

    auto gat_smem = [](int Fin) -> size_t {
        // xf + xb + wT + hT + att + sI + sJ + hp + adj double buffer
        return (size_t)Fin * (1024 + 512 + 128)
             + 32768 + 16384 + 1024 + 1024 + 8192 + 65536;
    };
    auto lstm_smem = [](int Din) -> size_t {
        // wf + wiT + whT + hb + bias + cst + gst
        return (size_t)Din * (1024 + 512)
             + 32768 + 2048 + 1024 + 4096 + 16384;
    };

    // GAT layer 0 (Fin=32) and layer 1 (Fin=64)
    gat_kernel<32><<<256, 256, gat_smem(32), stream>>>(x,  adj, gat0_W, gat0_a, gat0_g, gat0_b, g1);
    gat_kernel<64><<<256, 256, gat_smem(64), stream>>>(g1, adj, gat1_W, gat1_a, gat1_g, gat1_b, g2);

    // residual + gate + transpose into LSTM sequence layout
    comb_kernel<<<(4 * 64 * 256 * 64) / 256, 256, 0, stream>>>(x, g2, res_W, res_b, alpha, seqbf);

    // BiLSTM layer 0 (Din=64) writes o1; layer 1 (Din=128) writes final hidden
    lstm_kernel<64> <<<dim3(64, 2), 256, lstm_smem(64),  stream>>>(seqbf, l0_Wih, l0_Whh, l0_bih, l0_bhh, o1bf, nullptr);
    lstm_kernel<128><<<dim3(64, 2), 256, lstm_smem(128), stream>>>(o1bf,  l1_Wih, l1_Whh, l1_bih, l1_bhh, nullptr, hlast);

    // final FC
    fc_kernel<<<4, 256, 0, stream>>>(hlast, fc_W, fc_b, (float*)d_out);
}
